// CSILoss_87454124081308
// MI455X (gfx1250) — matrix operand order repaired and checked by reading, not compiled, not run on hardware
//
#include <hip/hip_runtime.h>
#include <hip/hip_bf16.h>

// ---------------------------------------------------------------------------
// CSILoss for MI455X (gfx1250): fused normalized-Gram + streaming softmax
// using v_wmma_f32_16x16x32_f16, async global->LDS B-tile staging, wave32.
// Bounded-logit trick: logits = 4*cos in [-4,4] => fixed softmax max = 4.0.
// ---------------------------------------------------------------------------

typedef __attribute__((ext_vector_type(16))) _Float16 v16h;
typedef __attribute__((ext_vector_type(8)))  _Float16 v8h;
typedef __attribute__((ext_vector_type(8)))  float    v8f;

typedef __attribute__((address_space(3))) _Float16 lds_h;

#define Bn 8192     // batch
#define Dn 256      // feature dim
#define KC (Dn/32)  // K-chunks of 32 per WMMA
#define WAVES 4     // waves per workgroup in the GEMM kernel
#define EPSN 1e-8f

// ---------------------------------------------------------------------------
// Kernel 0: zero the two scalar accumulators in workspace
// ---------------------------------------------------------------------------
__global__ void csil_init(float* __restrict__ acc) {
    if (threadIdx.x < 2) acc[threadIdx.x] = 0.0f;
}

// ---------------------------------------------------------------------------
// Kernel 1: row-normalize z (f32) -> zn (f16). One wave per row.
// ---------------------------------------------------------------------------
__global__ __launch_bounds__(256) void csil_normalize(
        const float* __restrict__ z, _Float16* __restrict__ zn) {
    const int row  = blockIdx.x * (blockDim.x >> 5) + (threadIdx.x >> 5);
    const int lane = threadIdx.x & 31;
    const float* zr = z + (size_t)row * Dn;

    float vals[Dn / 32];
    float ss = 0.0f;
#pragma unroll
    for (int i = 0; i < Dn / 32; ++i) {
        float v = zr[lane + 32 * i];
        vals[i] = v;
        ss += v * v;
    }
#pragma unroll
    for (int off = 16; off >= 1; off >>= 1)
        ss += __shfl_xor(ss, off, 32);

    float nrm = fmaxf(sqrtf(ss), EPSN);
    float s   = 1.0f / nrm;

    _Float16* o = zn + (size_t)row * Dn;
#pragma unroll
    for (int i = 0; i < Dn / 32; ++i)
        o[lane + 32 * i] = (_Float16)(vals[i] * s);
}

// ---------------------------------------------------------------------------
// Kernel 2: streaming contrastive softmax over the 8192x8192 Gram matrix.
// Each wave owns a 16-row M stripe; B tiles (16x256 f16, 8KB) staged in LDS
// via global_load_async_to_lds_b128 and shared by the 4 waves of the WG.
// ---------------------------------------------------------------------------
__global__ __launch_bounds__(WAVES * 32) void csil_contrastive(
        const _Float16* __restrict__ zn, float* __restrict__ c_accum) {
    __shared__ __align__(16) _Float16 Bs[16 * Dn];   // 8 KB B tile

    const int tid   = threadIdx.x;
    const int wave  = tid >> 5;
    const int lane  = tid & 31;
    const int l16   = lane & 15;
    const int lhi   = lane >> 4;                      // 0 or 1 (lane half)
    const int mbase = blockIdx.x * (WAVES * 16) + wave * 16;

    // ---- Load A fragments for this wave's 16 rows (all K chunks) ----------
    // ISA 16-bit A 16x32 layout: lane half lhi, elems e<8 -> K=lhi*8+e,
    // elems e>=8 -> K=16+lhi*8+(e-8).
    const _Float16* arow = zn + (size_t)(mbase + l16) * Dn;
    v16h a[KC];
#pragma unroll
    for (int kc = 0; kc < KC; ++kc) {
        v8h lo = *(const v8h*)(arow + kc * 32 + lhi * 8);
        v8h hi = *(const v8h*)(arow + kc * 32 + 16 + lhi * 8);
#pragma unroll
        for (int e = 0; e < 8; ++e) { a[kc][e] = lo[e]; a[kc][e + 8] = hi[e]; }
    }

    // ---- Streaming softmax state (fixed max = 4.0): 8 rows per lane -------
    float rsum[8], pos[8];
#pragma unroll
    for (int r = 0; r < 8; ++r) { rsum[r] = 0.0f; pos[r] = 0.0f; }

    // Cooperative B-tile staging: 128 lanes x 4 async b128 = 8 KB
    const int brow = tid >> 3;                 // 0..15
    const int bcol = (tid & 7) * 32;           // halves within row (0..224)
    const unsigned ldsbase =
        (unsigned)(unsigned long long)(lds_h*)&Bs[brow * Dn + bcol];

    const int NT = Bn / 16;
    for (int nt = 0; nt < NT; ++nt) {
        const int nbase = nt * 16;

        // Stage B tile: zn rows [nbase, nbase+16) -> LDS (async copy path)
        {
            const _Float16* gsrc = zn + (size_t)(nbase + brow) * Dn + bcol;
#pragma unroll
            for (int q = 0; q < 4; ++q) {
                asm volatile("global_load_async_to_lds_b128 %0, %1, off offset:%2"
                             :: "v"(ldsbase + q * 16), "v"(gsrc), "i"(q * 16)
                             : "memory");
            }
            if (nt + 1 < NT)   // hint next tile into cache (global_prefetch_b8)
                __builtin_prefetch((const void*)(gsrc + 16 * Dn), 0, 1);
            asm volatile("s_wait_asynccnt 0x0" ::: "memory");
        }
        __syncthreads();

        // ---- 16x16 logit tile via 8x v_wmma_f32_16x16x32_f16 --------------
        // B 32x16 layout: lane column n=l16, e<8 -> K=lhi*8+e, e>=8 -> K=16+lhi*8+(e-8).
        v8f c = {};
#pragma unroll
        for (int half = 0; half < 2; ++half) {
            v16h bf[KC / 2];
#pragma unroll
            for (int j = 0; j < KC / 2; ++j) {
                const int kc = half * (KC / 2) + j;
                const _Float16* bp = Bs + l16 * Dn + kc * 32 + lhi * 8;
                v8h b0 = *(const v8h*)(bp);
                v8h b1 = *(const v8h*)(bp + 16);
#pragma unroll
                for (int e = 0; e < 8; ++e) { bf[j][e] = b0[e]; bf[j][e + 8] = b1[e]; }
            }
#pragma unroll
            for (int j = 0; j < KC / 2; ++j) {
                c = __builtin_amdgcn_wmma_f32_16x16x32_f16(
                        false, a[half * (KC / 2) + j], false, bf[j],
                        (short)0, c, false, false);
            }
        }
        __syncthreads();

        // ---- Softmax accumulate; C layout: m = mbase+lhi*8+r, n = nbase+l16
        const int nglob = nbase + l16;
        if (nbase == mbase) {
#pragma unroll
            for (int r = 0; r < 8; ++r) {
                const int mg = mbase + lhi * 8 + r;
                float xs = fmaf(4.0f, c[r], -4.0f);     // logit - 4
                if (nglob == (mg ^ 1)) pos[r] = xs + 4.0f;
                if (nglob != mg) rsum[r] += __expf(xs); // mask diagonal
            }
        } else {
#pragma unroll
            for (int r = 0; r < 8; ++r)
                rsum[r] += __expf(fmaf(4.0f, c[r], -4.0f));
        }
    }

    // ---- Sum rsum and pos across the 16 n-lanes of each half --------------
#pragma unroll
    for (int r = 0; r < 8; ++r) {
#pragma unroll
        for (int off = 1; off < 16; off <<= 1) {
            rsum[r] += __shfl_xor(rsum[r], off, 32);
            pos[r]  += __shfl_xor(pos[r],  off, 32);
        }
    }

    if (l16 == 0) {
        float s = 0.0f;
#pragma unroll
        for (int r = 0; r < 8; ++r)
            s += (4.0f + logf(rsum[r])) - pos[r];       // lse - pos
        atomicAdd(c_accum, s);
    }
}

// ---------------------------------------------------------------------------
// Kernel 3: rotation cross-entropy over [B,4] logits, labels % 4
// ---------------------------------------------------------------------------
__global__ __launch_bounds__(256) void csil_rotation(
        const float* __restrict__ rp, const long long* __restrict__ labels,
        float* __restrict__ r_accum) {
    const int i = blockIdx.x * blockDim.x + threadIdx.x;
    float nll = 0.0f;
    if (i < Bn) {
        float4 v = ((const float4*)rp)[i];
        float mx = fmaxf(fmaxf(v.x, v.y), fmaxf(v.z, v.w));
        float se = __expf(v.x - mx) + __expf(v.y - mx) +
                   __expf(v.z - mx) + __expf(v.w - mx);
        float lse = mx + logf(se);
        int lab = (int)(labels[i] & 3);
        float sel = (lab == 0) ? v.x : (lab == 1) ? v.y : (lab == 2) ? v.z : v.w;
        nll = lse - sel;
    }
#pragma unroll
    for (int off = 16; off >= 1; off >>= 1)
        nll += __shfl_xor(nll, off, 32);
    if ((threadIdx.x & 31) == 0) atomicAdd(r_accum, nll);
}

// ---------------------------------------------------------------------------
// Kernel 4: finalize — means + lambda combine
// ---------------------------------------------------------------------------
__global__ void csil_finalize(const float* __restrict__ acc, float* __restrict__ out) {
    out[0] = acc[0] / (float)Bn + acc[1] / (float)Bn;   // LAMDA = 1
}

// ---------------------------------------------------------------------------
extern "C" void kernel_launch(void* const* d_in, const int* in_sizes, int n_in,
                              void* d_out, int out_size, void* d_ws, size_t ws_size,
                              hipStream_t stream) {
    const float*     z      = (const float*)d_in[0];      // [8192, 256] f32
    const float*     rp     = (const float*)d_in[1];      // [8192, 4]   f32
    const long long* labels = (const long long*)d_in[2];  // [8192]      i64
    float*           out    = (float*)d_out;              // scalar f32

    // Workspace layout: acc[2] floats at offset 0, zn (f16, 4MB) at +256B
    float*     acc = (float*)d_ws;
    _Float16*  zn  = (_Float16*)((char*)d_ws + 256);

    csil_init<<<1, 32, 0, stream>>>(acc);
    csil_normalize<<<Bn / 8, 256, 0, stream>>>(z, zn);
    csil_contrastive<<<Bn / (WAVES * 16), WAVES * 32, 0, stream>>>(zn, acc + 0);
    csil_rotation<<<Bn / 256, 256, 0, stream>>>(rp, labels, acc + 1);
    csil_finalize<<<1, 1, 0, stream>>>(acc, out);
}
